// HMM_32968168964717
// MI455X (gfx1250) — matrix operand order repaired and checked by reading, not compiled, not run on hardware
//
#include <hip/hip_runtime.h>
#include <hip/hip_bf16.h>

typedef __attribute__((ext_vector_type(16))) _Float16 v16h;
typedef __attribute__((ext_vector_type(8)))  _Float16 v8h;
typedef __attribute__((ext_vector_type(8)))  float    v8f;

#define HMM_B    64
#define HMM_TMAX 2048
#define HMM_N    256
#define HMM_M    1024

// workspace layout (bytes):
//   [0,       1024)            pi_prob  f32[256]           softmax(pi)
//   [4096,    4096+128K)       Pf16     f16[256*256]       P[i*N+k] = p(state i | state k), col-stochastic
//   [135168,  135168+1M)       Et       f32[1024*256]      Et[m*256+i] = p(obs m | state i)  (transposed)
#define WS_PI    0
#define WS_P     4096
#define WS_ET    135168

// ---------------- preprocessing ----------------

__global__ void hmm_prep_pi(const float* __restrict__ pi, float* __restrict__ pi_prob) {
    __shared__ float red[256];
    int i = threadIdx.x;
    float v = pi[i];
    red[i] = v; __syncthreads();
    for (int s = 128; s > 0; s >>= 1) { if (i < s) red[i] = fmaxf(red[i], red[i + s]); __syncthreads(); }
    float m = red[0]; __syncthreads();
    float e = expf(v - m);
    red[i] = e; __syncthreads();
    for (int s = 128; s > 0; s >>= 1) { if (i < s) red[i] += red[i + s]; __syncthreads(); }
    pi_prob[i] = e / red[0];
}

// column softmax of trans (axis=0): block = column k, thread = row i
__global__ void hmm_prep_trans(const float* __restrict__ trans, _Float16* __restrict__ Pf16) {
    __shared__ float red[256];
    int k = blockIdx.x, i = threadIdx.x;
    float v = trans[i * HMM_N + k];
    red[i] = v; __syncthreads();
    for (int s = 128; s > 0; s >>= 1) { if (i < s) red[i] = fmaxf(red[i], red[i + s]); __syncthreads(); }
    float m = red[0]; __syncthreads();
    float e = expf(v - m);
    red[i] = e; __syncthreads();
    for (int s = 128; s > 0; s >>= 1) { if (i < s) red[i] += red[i + s]; __syncthreads(); }
    Pf16[i * HMM_N + k] = (_Float16)(e / red[0]);
}

// row softmax of emis (axis=1): block = state i, 256 threads x 4 symbols; store transposed
__global__ void hmm_prep_emis(const float* __restrict__ emis, float* __restrict__ Et) {
    __shared__ float red[256];
    int i = blockIdx.x, t = threadIdx.x;
    float v[4];
    float mx = -INFINITY;
    #pragma unroll
    for (int j = 0; j < 4; ++j) { v[j] = emis[i * HMM_M + t + 256 * j]; mx = fmaxf(mx, v[j]); }
    red[t] = mx; __syncthreads();
    for (int s = 128; s > 0; s >>= 1) { if (t < s) red[t] = fmaxf(red[t], red[t + s]); __syncthreads(); }
    float m = red[0]; __syncthreads();
    float ps = 0.f;
    #pragma unroll
    for (int j = 0; j < 4; ++j) { v[j] = expf(v[j] - m); ps += v[j]; }
    red[t] = ps; __syncthreads();
    for (int s = 128; s > 0; s >>= 1) { if (t < s) red[t] += red[t + s]; __syncthreads(); }
    float inv = 1.0f / red[0];
    #pragma unroll
    for (int j = 0; j < 4; ++j) Et[(t + 256 * j) * HMM_N + i] = v[j] * inv;
}

// ---------------- forward recursion ----------------
// grid = 4 blocks (16 batches each), block = 512 threads = 16 wave32.
// wave w owns output rows [w*16, w*16+16); lane&15 = local batch column.
// Double-buffered by step parity: alpha (f16), emission stage, scale slots.
// Step t reads alphaBuf[t&1] / ebuf[t&1] / s_lds[t&1], writes alphaBuf[1-(t&1)],
// and pre-stages emission t+1 into ebuf[1-(t&1)] during the WMMA phase.

#define AH_STRIDE 264   // halves  (16*264*2 = 8448 B per buffer)
#define EL_STRIDE 260   // floats  (16*260*4 = 16640 B per buffer)

__global__ void __launch_bounds__(512)
hmm_forward(const int* __restrict__ x, const int* __restrict__ T,
            const float* __restrict__ pi_prob, const _Float16* __restrict__ Pf16,
            const float* __restrict__ Et, float* __restrict__ out)
{
    __shared__ _Float16 alphaBuf[2 * 16 * AH_STRIDE];
    __shared__ float    ebuf[2 * 16 * EL_STRIDE];
    __shared__ float    s_lds[2 * 16];

    const int tid   = threadIdx.x;
    const int wave  = tid >> 5;
    const int lane  = tid & 31;
    const int n     = lane & 15;     // local batch column (and this thread's logsum owner)
    const int hi    = lane >> 4;     // half-wave select
    const int bbase = blockIdx.x * 16;

    // --- load this wave's A fragments (16 rows of P, all K=256) into registers ---
    // 16-bit A layout: lanes 0-15 hold M=lane; lanes 16-31 hold M=lane-16.
    //   VGPR 0..3 : K = {0..7}+hi*8 ; VGPR 4..7 : K = {16..23}+hi*8  (per 32-wide chunk)
    const int mrow = wave * 16 + n;
    const _Float16* prow = Pf16 + mrow * HMM_N;
    v16h a[8];
    #pragma unroll
    for (int c = 0; c < 8; ++c) {
        const int k0 = c * 32 + hi * 8;
        v8h lo = *(const v8h*)(prow + k0);
        v8h hh = *(const v8h*)(prow + k0 + 16);
        #pragma unroll
        for (int j = 0; j < 8; ++j) { a[c][j] = lo[j]; a[c][8 + j] = hh[j]; }
    }

    const int Tb = T[bbase + n];

    // per-step emission staging: thread (tid>>5) = batch col, lane*8 = state offset
    const int    bj   = tid >> 5;                       // 0..15
    const int    i0   = lane * 8;                       // 0..248
    const int*   xrow = x + (bbase + bj) * HMM_TMAX;
    auto stage_e = [&](int t, int buf) {
        const int m = xrow[t];
        const float4* src = (const float4*)(Et + m * HMM_N + i0);
        float4* dst = (float4*)(ebuf + buf * (16 * EL_STRIDE) + bj * EL_STRIDE + i0);
        dst[0] = src[0]; dst[1] = src[1];
    };

    // B-fragment load from alpha buffer: lane holds column n;
    // K = c*32 + {0..7}+hi*8, then c*32 + {16..23}+hi*8 (matches A layout)
    auto load_b = [&](const _Float16* ab, int c) -> v16h {
        const _Float16* bp = ab + n * AH_STRIDE + c * 32 + hi * 8;
        v8h b0 = *(const v8h*)bp;
        v8h b1 = *(const v8h*)(bp + 16);
        v16h bf;
        #pragma unroll
        for (int j = 0; j < 8; ++j) { bf[j] = b0[j]; bf[8 + j] = b1[j]; }
        return bf;
    };

    float logsum = 0.f;
    float val[8];

    // ---- t = 0 : alpha0[i,b] = pi_prob[i] * E[i, x[b,0]] ----
    if (tid < 16) { s_lds[tid] = 0.f; s_lds[16 + tid] = 0.f; }
    stage_e(0, 0);
    __syncthreads();
    {
        stage_e(1, 1);                               // prefetch emission for t=1
        float part = 0.f;
        #pragma unroll
        for (int r = 0; r < 8; ++r) {
            const int i = wave * 16 + r + hi * 8;    // D layout rows
            float vv = pi_prob[i] * ebuf[n * EL_STRIDE + i];
            val[r] = vv; part += vv;
        }
        atomicAdd(&s_lds[n], part);
    }
    __syncthreads();
    {
        const float s = s_lds[n];
        const float inv = __builtin_amdgcn_rcpf(s);
        #pragma unroll
        for (int r = 0; r < 8; ++r) {
            const int i = wave * 16 + r + hi * 8;
            alphaBuf[(16 * AH_STRIDE) + n * AH_STRIDE + i] = (_Float16)(val[r] * inv);  // buf 1 = t=1 input
        }
        logsum += __logf(s);
        if (tid < 16 && Tb == 1) out[bbase + tid] = logsum;
    }

    // ---- t = 1 .. TMAX-1 : two barriers per step ----
    for (int t = 1; t < HMM_TMAX; ++t) {
        const int p  = t & 1;
        const int q  = p ^ 1;
        const _Float16* aIn  = alphaBuf + p * (16 * AH_STRIDE);
        _Float16*       aOut = alphaBuf + q * (16 * AH_STRIDE);
        const float*    eIn  = ebuf + p * (16 * EL_STRIDE);

        __syncthreads();                              // barrier A: prev writes visible

        // overlap with WMMA: stage next emission, re-zero next scale slots
        if (t + 1 < HMM_TMAX) stage_e(t + 1, q);
        if (tid < 16) s_lds[q * 16 + tid] = 0.f;

        // D(16x16) = sum_c A_tile(16x32) * B_tile(32x16): alpha' = P * alpha
        v8f acc = {};
        v16h bcur = load_b(aIn, 0);
        #pragma unroll
        for (int c = 0; c < 8; ++c) {
            v16h bnext;
            if (c < 7) bnext = load_b(aIn, c + 1);    // prefetch next K-chunk
            acc = __builtin_amdgcn_wmma_f32_16x16x32_f16(
                      false, a[c], false, bcur, (short)0, acc, false, false);
            bcur = bnext;
        }

        // emission multiply + per-column scale
        float part = 0.f;
        #pragma unroll
        for (int r = 0; r < 8; ++r) {
            const int i = wave * 16 + r + hi * 8;
            float vv = acc[r] * eIn[n * EL_STRIDE + i];
            val[r] = vv; part += vv;
        }
        atomicAdd(&s_lds[p * 16 + n], part);
        __syncthreads();                              // barrier B: sums + staging complete

        const float s = s_lds[p * 16 + n];
        const float inv = __builtin_amdgcn_rcpf(s);
        #pragma unroll
        for (int r = 0; r < 8; ++r) {
            const int i = wave * 16 + r + hi * 8;
            aOut[n * AH_STRIDE + i] = (_Float16)(val[r] * inv);
        }
        logsum += __logf(s);
        if (tid < 16 && t == Tb - 1) out[bbase + tid] = logsum;
    }
}

// ---------------- launcher ----------------

extern "C" void kernel_launch(void* const* d_in, const int* in_sizes, int n_in,
                              void* d_out, int out_size, void* d_ws, size_t ws_size,
                              hipStream_t stream) {
    (void)in_sizes; (void)n_in; (void)out_size; (void)ws_size;
    const int*   x     = (const int*)  d_in[0];
    const int*   T     = (const int*)  d_in[1];
    const float* pi    = (const float*)d_in[2];
    const float* trans = (const float*)d_in[3];
    const float* emis  = (const float*)d_in[4];
    float* out = (float*)d_out;

    char* ws = (char*)d_ws;
    float*    pi_prob = (float*)   (ws + WS_PI);
    _Float16* Pf16    = (_Float16*)(ws + WS_P);
    float*    Et      = (float*)   (ws + WS_ET);

    hmm_prep_pi   <<<1,   256, 0, stream>>>(pi, pi_prob);
    hmm_prep_trans<<<256, 256, 0, stream>>>(trans, Pf16);
    hmm_prep_emis <<<256, 256, 0, stream>>>(emis, Et);
    hmm_forward   <<<4,   512, 0, stream>>>(x, T, pi_prob, Pf16, Et, out);
}
